// MoEAdaptorsLinear_13649406067317
// MI455X (gfx1250) — matrix operand order repaired and checked by reading, not compiled, not run on hardware
//
#include <hip/hip_runtime.h>

typedef __attribute__((ext_vector_type(16))) __bf16 v16bf;
typedef __attribute__((ext_vector_type(8)))  __bf16 v8bf;
typedef __attribute__((ext_vector_type(4)))  __bf16 v4bf;
typedef __attribute__((ext_vector_type(8)))  float  v8f;
typedef __attribute__((ext_vector_type(4)))  float  v4f;

#define N_TOK   8192
#define DIM     2048
#define RNK     64
#define NEXP    8
#define SCALE_F 4.0f   // 64 / 16

// ---------------------------------------------------------------- utilities

__global__ void zero_counts_kernel(int* __restrict__ cnt) {
    if (threadIdx.x < NEXP) cnt[threadIdx.x] = 0;
}

__global__ void f32_to_bf16_kernel(const float* __restrict__ src,
                                   __bf16* __restrict__ dst, int n4) {
    int i = blockIdx.x * blockDim.x + threadIdx.x;
    if (i >= n4) return;
    v4f v = reinterpret_cast<const v4f*>(src)[i];
    v4bf o;
    o[0] = (__bf16)v[0]; o[1] = (__bf16)v[1];
    o[2] = (__bf16)v[2]; o[3] = (__bf16)v[3];
    reinterpret_cast<v4bf*>(dst)[i] = o;
}

// ------------------------------------------------- gate + x->bf16 (fused)
// One wave (32 lanes) per token. Wg (8x2048 f32 = 64KB) staged in LDS.

__global__ __launch_bounds__(256)
void gate_kernel(const float* __restrict__ x, const float* __restrict__ wg,
                 __bf16* __restrict__ xb, float* __restrict__ coef,
                 int* __restrict__ cnt, int* __restrict__ tok_list) {
    __shared__ float wgl[NEXP * DIM];
    for (int i = threadIdx.x; i < NEXP * DIM / 4; i += blockDim.x)
        reinterpret_cast<v4f*>(wgl)[i] = reinterpret_cast<const v4f*>(wg)[i];
    __syncthreads();

    const int wave = threadIdx.x >> 5;
    const int lane = threadIdx.x & 31;
    const int tok  = blockIdx.x * 8 + wave;

    const float* xr  = x  + (size_t)tok * DIM;
    __bf16*      xbr = xb + (size_t)tok * DIM;

    float acc[NEXP];
#pragma unroll
    for (int e = 0; e < NEXP; ++e) acc[e] = 0.f;

#pragma unroll
    for (int c = 0; c < 16; ++c) {
        const int d4 = c * 32 + lane;
        v4f xv = reinterpret_cast<const v4f*>(xr)[d4];
        v4bf xo;
        xo[0] = (__bf16)xv[0]; xo[1] = (__bf16)xv[1];
        xo[2] = (__bf16)xv[2]; xo[3] = (__bf16)xv[3];
        reinterpret_cast<v4bf*>(xbr)[d4] = xo;
#pragma unroll
        for (int e = 0; e < NEXP; ++e) {
            v4f wv = reinterpret_cast<const v4f*>(wgl + e * DIM)[d4];
            acc[e] = fmaf(xv[0], wv[0],
                     fmaf(xv[1], wv[1],
                     fmaf(xv[2], wv[2],
                     fmaf(xv[3], wv[3], acc[e]))));
        }
    }
    // wave32 reduction
#pragma unroll
    for (int off = 16; off > 0; off >>= 1)
#pragma unroll
        for (int e = 0; e < NEXP; ++e)
            acc[e] += __shfl_xor(acc[e], off, 32);

    if (lane == 0) {
        float m = acc[0]; int g = 0;
#pragma unroll
        for (int e = 1; e < NEXP; ++e)
            if (acc[e] > m) { m = acc[e]; g = e; }
        float s = 0.f;
#pragma unroll
        for (int e = 0; e < NEXP; ++e) s += __expf(acc[e] - m);
        coef[tok] = SCALE_F / s;           // SCALING * prob_sel  (p_sel = 1/s)
        int pos = atomicAdd(&cnt[g], 1);
        tok_list[g * N_TOK + pos] = tok;
    }
}

// --------------------------------------- fused LoRA: h = Xg*A_e^T, y = h*B_e^T
// Block = 128 threads (4 waves). blockIdx.x = 16-token tile, blockIdx.y = expert.

__global__ __launch_bounds__(128)
void moe_lora_kernel(const __bf16* __restrict__ xb, const __bf16* __restrict__ ab,
                     const __bf16* __restrict__ bb, const float* __restrict__ coef,
                     const int* __restrict__ cnt, const int* __restrict__ tok_list,
                     float* __restrict__ out) {
    const int e  = blockIdx.y;
    const int mt = blockIdx.x;
    const int cn = cnt[e];
    if (mt * 16 >= cn) return;

    __shared__ int    tok_s[16];
    __shared__ float  cf_s[16];
    __shared__ __align__(16) float  hf[16 * RNK];
    __shared__ __align__(16) __bf16 hb[16 * RNK];

    const int tid  = threadIdx.x;
    const int lane = tid & 31;
    const int wave = tid >> 5;
    const int col  = lane & 15;     // matrix row/col index within a 16-wide tile
    const int hi   = lane >> 4;     // half-wave select

    if (tid < 16) {
        int i = mt * 16 + tid;
        if (i >= cn) i = cn - 1;                 // clamp padded rows
        int t = tok_list[e * N_TOK + i];
        tok_s[tid] = t;
        cf_s[tid]  = coef[t];
    }
    for (int i = tid; i < 16 * RNK; i += 128) hf[i] = 0.f;
    __syncthreads();

    // ---------------- phase 1: h[16 x 64] = gather(X)[16 x 2048] * A_e^T
    const __bf16* xrow = xb + (size_t)tok_s[col] * DIM;  // A-frag row for this lane
    v8f acc[4] = {};
    for (int ks = 0; ks < 16; ++ks) {
        const int k0 = (wave * 16 + ks) * 32;
        // A fragment (16-bit 16x32): chunks {c0, c0+16}, c0 = k0 + (hi?8:0)
        const int c0 = k0 + (hi ? 8 : 0);
        union { v16bf v; v8bf h[2]; } ua;
        ua.h[0] = *reinterpret_cast<const v8bf*>(xrow + c0);
        ua.h[1] = *reinterpret_cast<const v8bf*>(xrow + c0 + 16);
        // B fragment (32x16): chunks {kb, kb+16}, kb = k0 + (hi?8:0)
        const int kb = k0 + (hi ? 8 : 0);
#pragma unroll
        for (int rt = 0; rt < 4; ++rt) {
            const __bf16* brow = ab + ((size_t)e * RNK + rt * 16 + col) * DIM;
            union { v16bf v; v8bf h[2]; } ub;
            ub.h[0] = *reinterpret_cast<const v8bf*>(brow + kb);
            ub.h[1] = *reinterpret_cast<const v8bf*>(brow + kb + 16);
            acc[rt] = __builtin_amdgcn_wmma_f32_16x16x32_bf16(
                false, ua.v, false, ub.v, (short)0, acc[rt], false, false);
        }
    }
    // reduce the 4 waves' K-partials in LDS
#pragma unroll
    for (int rt = 0; rt < 4; ++rt)
#pragma unroll
        for (int vr = 0; vr < 8; ++vr) {
            const int m = vr + 8 * hi;                       // C layout row
            atomicAdd(&hf[m * RNK + rt * 16 + col], acc[rt][vr]);
        }
    __syncthreads();
    for (int i = tid; i < 16 * RNK; i += 128) hb[i] = (__bf16)hf[i];
    __syncthreads();

    // ---------------- phase 2: y[16 x 2048] = h[16 x 64] * B_e^T, scaled
    v16bf ha[2];
#pragma unroll
    for (int s = 0; s < 2; ++s) {
        const int c0 = s * 32 + (hi ? 8 : 0);
        union { v16bf v; v8bf h[2]; } u;
        u.h[0] = *reinterpret_cast<const v8bf*>(&hb[col * RNK + c0]);
        u.h[1] = *reinterpret_cast<const v8bf*>(&hb[col * RNK + c0 + 16]);
        ha[s] = u.v;
    }
    float  cfl[8];
    float* orow[8];
    bool   val[8];
#pragma unroll
    for (int vr = 0; vr < 8; ++vr) {
        const int m = vr + 8 * hi;
        cfl[vr]  = cf_s[m];
        val[vr]  = (mt * 16 + m) < cn;
        orow[vr] = out + (size_t)tok_s[m] * DIM;
    }
    for (int nt = wave; nt < DIM / 16; nt += 4) {
        const __bf16* bbase = bb + ((size_t)e * DIM + nt * 16 + col) * RNK;
        v8f acc2 = {};
#pragma unroll
        for (int s = 0; s < 2; ++s) {
            const int kb = s * 32 + (hi ? 8 : 0);
            union { v16bf v; v8bf h[2]; } u;
            u.h[0] = *reinterpret_cast<const v8bf*>(bbase + kb);
            u.h[1] = *reinterpret_cast<const v8bf*>(bbase + kb + 16);
            acc2 = __builtin_amdgcn_wmma_f32_16x16x32_bf16(
                false, ha[s], false, u.v, (short)0, acc2, false, false);
        }
        const int o = nt * 16 + col;
#pragma unroll
        for (int vr = 0; vr < 8; ++vr)
            if (val[vr]) orow[vr][o] = acc2[vr] * cfl[vr];
    }
}

// ---------------------------------------------------------------- launcher

extern "C" void kernel_launch(void* const* d_in, const int* in_sizes, int n_in,
                              void* d_out, int out_size, void* d_ws, size_t ws_size,
                              hipStream_t stream) {
    const float* x  = (const float*)d_in[0];   // [N, D]
    const float* A  = (const float*)d_in[1];   // [E, R, D]
    const float* Bw = (const float*)d_in[2];   // [E, D, R]
    const float* Wg = (const float*)d_in[3];   // [E, D]
    float* out = (float*)d_out;

    char* ws = (char*)d_ws;
    size_t off = 0;
    auto take = [&](size_t bytes) { void* p = ws + off; off = (off + bytes + 63) & ~(size_t)63; return p; };

    __bf16* xb   = (__bf16*)take((size_t)N_TOK * DIM * 2);     // 32 MB
    __bf16* abuf = (__bf16*)take((size_t)NEXP * RNK * DIM * 2);//  2 MB
    __bf16* bbuf = (__bf16*)take((size_t)NEXP * DIM * RNK * 2);//  2 MB
    float*  coef = (float*)take((size_t)N_TOK * 4);
    int*    cnt  = (int*)take(NEXP * 4);
    int*    tokl = (int*)take((size_t)NEXP * N_TOK * 4);

    zero_counts_kernel<<<1, 32, 0, stream>>>(cnt);

    const int nA4 = NEXP * RNK * DIM / 4;
    f32_to_bf16_kernel<<<(nA4 + 255) / 256, 256, 0, stream>>>(A,  abuf, nA4);
    f32_to_bf16_kernel<<<(nA4 + 255) / 256, 256, 0, stream>>>(Bw, bbuf, nA4);

    gate_kernel<<<N_TOK / 8, 256, 0, stream>>>(x, Wg, xb, coef, cnt, tokl);

    moe_lora_kernel<<<dim3(N_TOK / 16, NEXP), 128, 0, stream>>>(
        xb, abuf, bbuf, coef, cnt, tokl, out);
}
